// Model_28793460752976
// MI455X (gfx1250) — matrix operand order repaired and checked by reading, compiled, not run on hardware
//
#include <hip/hip_runtime.h>

// ---------------- Problem constants (from reference) ----------------
#define BB      8
#define LL      2048
#define ENCIN   32
#define DMODEL  512
#define DINNER  1024
#define DSTATE  16
#define DTRANK  32
#define NTOK    (BB * LL)          // 16384 tokens

typedef _Float16 half_t;
typedef _Float16 v16h __attribute__((ext_vector_type(16)));
typedef float    v8f  __attribute__((ext_vector_type(8)));
typedef unsigned int u32x4 __attribute__((ext_vector_type(4)));
typedef unsigned int u32x8 __attribute__((ext_vector_type(8)));

// =====================================================================
// Generic f16 WMMA GEMM:  C[M,N](f32) = A[M,lda](f16) * W[N,K](f16)^T
// Block: 256 threads (8 wave32), tile 128(M) x 64(N), K step 32.
// A tile fed by per-lane GLOBAL_LOAD_ASYNC_TO_LDS_B128 (ASYNCcnt).
// B tile fed by one Tensor Data Mover tensor_load_to_lds (TENSORcnt).
// Wave computes 32x32 => 2x2 of v_wmma_f32_16x16x32_f16.
// =====================================================================
#define TILE_M 128
#define TILE_N 64
#define TILE_K 32
#define ASTR   48   // A LDS row stride in halves (96B: 16B-aligned chunks, bank-skewed)

__global__ __launch_bounds__(256) void k_gemm_f16_wmma(
    const half_t* __restrict__ A, int lda,      // [M, lda] row-major
    const half_t* __restrict__ W,               // [N, K]  row-major
    float* __restrict__ C, int ldc,             // [M, ldc]
    int M, int N, int K)
{
  __shared__ half_t As[TILE_M * ASTR];
  __shared__ half_t Bs[TILE_N * TILE_K];        // TDM writes tile linearly: stride 32 halves

  const int tid  = threadIdx.x;
  const int lane = tid & 31;
  const int wid  = tid >> 5;
  const int wm   = wid & 3;          // wave row (0..3) -> 32 rows each
  const int wn   = wid >> 2;         // wave col (0..1) -> 32 cols each
  const int m_blk = blockIdx.x * TILE_M;
  const int n_blk = blockIdx.y * TILE_N;
  const int lo   = lane & 15;
  const bool hi  = (lane >= 16);

  const unsigned lds_as = (unsigned)(unsigned long long)(&As[0]);
  const unsigned lds_bs = (unsigned)(unsigned long long)(&Bs[0]);

  v8f acc[2][2];
#pragma unroll
  for (int i = 0; i < 2; ++i)
#pragma unroll
    for (int j = 0; j < 2; ++j)
#pragma unroll
      for (int v = 0; v < 8; ++v) acc[i][j][v] = 0.0f;

  for (int k0 = 0; k0 < K; k0 += TILE_K) {
    __syncthreads();   // previous compute finished before LDS overwrite

    // ---- A tile: 128 x 32 halves, per-lane async global->LDS b128 copies
#pragma unroll
    for (int p = 0; p < 2; ++p) {
      int ch  = tid + p * 256;           // 0..511 chunks of 8 halves
      int row = ch >> 2;
      int cc  = ch & 3;
      unsigned lds = lds_as + (unsigned)((row * ASTR + cc * 8) * 2);
      unsigned long long ga = (unsigned long long)
          (A + (size_t)(m_blk + row) * (size_t)lda + (size_t)(k0 + cc * 8));
      asm volatile("global_load_async_to_lds_b128 %0, %1, off"
                   :: "v"(lds), "v"(ga) : "memory");
    }

    // ---- B tile: 64(N) x 32(K) halves via Tensor Data Mover, wave 0 only
    if (wid == 0) {
      unsigned long long ga = (unsigned long long)
          (W + (size_t)n_blk * (size_t)K + (size_t)k0);
      u32x4 g0;
      u32x8 g1;
      g0[0] = 1u;                                   // count=1, user descriptor
      g0[1] = lds_bs;                               // lds_addr
      g0[2] = (unsigned)ga;                         // global_addr[31:0]
      g0[3] = (unsigned)(ga >> 32) | (2u << 30);    // global_addr[56:32] | type=2
      g1[0] = (1u << 16);                           // data_size=2B, no flags
      g1[1] = ((unsigned)K & 0xFFFFu) << 16;        // tensor_dim0[15:0]
      g1[2] = ((unsigned)K >> 16) |
              (((unsigned)N & 0xFFFFu) << 16);      // tensor_dim0[31:16] | tensor_dim1[15:0]
      g1[3] = ((unsigned)N >> 16) |
              ((unsigned)TILE_K << 16);             // tensor_dim1[31:16] | tile_dim0=32
      g1[4] = (unsigned)TILE_N;                     // tile_dim1=64, tile_dim2=0
      g1[5] = (unsigned)K;                          // tensor_dim0_stride[31:0] = row pitch
      g1[6] = 0u;                                   // stride hi / dim1_stride lo
      g1[7] = 0u;
      asm volatile("tensor_load_to_lds %0, %1" :: "s"(g0), "s"(g1) : "memory");
      __builtin_amdgcn_s_wait_tensorcnt(0);
    }

    // ---- Prefetch next K tile of A (global_prefetch_b8 path)
    if (k0 + TILE_K < K) {
      __builtin_prefetch(
          (const void*)(A + (size_t)(m_blk + (tid >> 1)) * (size_t)lda +
                        (size_t)(k0 + TILE_K)), 0, 1);
    }

    asm volatile("s_wait_asynccnt 0" ::: "memory");
    __syncthreads();   // publish LDS tiles to all waves

    // ---- Build fragments per documented 16-bit layouts
    union Frag { uint4 u[2]; v16h h; };
    Frag af[2], bf[2];
#pragma unroll
    for (int mi = 0; mi < 2; ++mi) {
      // A 16x32: lanes 0-15 -> K 0..7 & 16..23 ; lanes 16-31 -> K 8..15 & 24..31
      int row   = wm * 32 + mi * 16 + lo;
      int cbase = hi ? 8 : 0;
      af[mi].u[0] = *reinterpret_cast<const uint4*>(&As[row * ASTR + cbase]);
      af[mi].u[1] = *reinterpret_cast<const uint4*>(&As[row * ASTR + cbase + 16]);
    }
#pragma unroll
    for (int ni = 0; ni < 2; ++ni) {
      // B 32x16: lanes 0-15 -> K 0..7 & 16..23 ; lanes 16-31 -> K 8..15 & 24..31 (N = lane%16)
      int row   = wn * 32 + ni * 16 + lo;
      int cbase = hi ? 8 : 0;
      bf[ni].u[0] = *reinterpret_cast<const uint4*>(&Bs[row * TILE_K + cbase]);
      bf[ni].u[1] = *reinterpret_cast<const uint4*>(&Bs[row * TILE_K + cbase + 16]);
    }
#pragma unroll
    for (int mi = 0; mi < 2; ++mi)
#pragma unroll
      for (int ni = 0; ni < 2; ++ni)
        acc[mi][ni] = __builtin_amdgcn_wmma_f32_16x16x32_f16(
            /*neg_a=*/false, af[mi].h, /*neg_b=*/false, bf[ni].h,
            /*c_mod=*/(short)0, acc[mi][ni],
            /*reuse_a=*/false, /*reuse_b=*/false);
  }

  // ---- Store C (f32 16x16 layout: VGPR v -> M = v + (hi?8:0), N = lane%16)
#pragma unroll
  for (int mi = 0; mi < 2; ++mi)
#pragma unroll
    for (int ni = 0; ni < 2; ++ni) {
      int col   = n_blk + wn * 32 + ni * 16 + lo;
      int rbase = m_blk + wm * 32 + mi * 16 + (hi ? 8 : 0);
#pragma unroll
      for (int v = 0; v < 8; ++v)
        C[(size_t)(rbase + v) * (size_t)ldc + col] = acc[mi][ni][v];
    }
}

// =====================================================================
// f32 -> f16 conversion
// =====================================================================
__global__ __launch_bounds__(256) void k_cvt_f16(
    const float* __restrict__ s, half_t* __restrict__ d, int n)
{
  int i = blockIdx.x * 256 + threadIdx.x;
  if (i < n) d[i] = (half_t)s[i];
}

// =====================================================================
// DataEmbedding: circular conv1d(k=3, 32->512) + temporal linear + pos emb
// =====================================================================
__global__ __launch_bounds__(256) void k_embed(
    const float* __restrict__ x_enc,     // [B,L,32]
    const float* __restrict__ x_mark,    // [B,L,4]
    const float* __restrict__ pos,       // [L,512]
    const float* __restrict__ convw,     // [3,32,512]
    const float* __restrict__ tempw,     // [512,4]
    half_t* __restrict__ xh)             // [B*L,512]
{
  const int bl = blockIdx.x;
  const int l  = bl & (LL - 1);
  const int b  = bl / LL;
  __shared__ float xe[96];
  __shared__ float mk[4];
  const int tid = threadIdx.x;
  const int lm1 = (l + LL - 1) & (LL - 1);
  const int lp1 = (l + 1) & (LL - 1);
  if (tid < 32)       xe[tid]      = x_enc[((size_t)b * LL + lm1) * ENCIN + tid];
  else if (tid < 64)  xe[tid]      = x_enc[((size_t)b * LL + l)   * ENCIN + (tid - 32)];
  else if (tid < 96)  xe[tid]      = x_enc[((size_t)b * LL + lp1) * ENCIN + (tid - 64)];
  else if (tid < 100) mk[tid - 96] = x_mark[((size_t)b * LL + l) * 4 + (tid - 96)];
  __syncthreads();
  for (int d = tid; d < DMODEL; d += 256) {
    float s = pos[(size_t)l * DMODEL + d];
#pragma unroll
    for (int k = 0; k < 3; ++k)
      for (int m = 0; m < ENCIN; ++m)
        s += xe[k * ENCIN + m] * convw[((size_t)(k * ENCIN + m)) * DMODEL + d];
#pragma unroll
    for (int m = 0; m < 4; ++m) s += mk[m] * tempw[d * 4 + m];
    xh[(size_t)bl * DMODEL + d] = (half_t)s;
  }
}

// =====================================================================
// Causal depthwise conv1d (k=4) on xm (= xz[...,:1024]) + bias + SiLU
// =====================================================================
__global__ __launch_bounds__(256) void k_dwconv_silu(
    const float* __restrict__ xz,        // [B*L,2048]
    const float* __restrict__ w,         // [4,1,1024]
    const float* __restrict__ bias,      // [1024]
    float* __restrict__ u,               // [B*L,1024]
    half_t* __restrict__ uh)             // [B*L,1024]
{
  const int bl = blockIdx.x;
  const int l  = bl & (LL - 1);
  const size_t brow = (size_t)(bl - l);  // b*L
  for (int d = threadIdx.x; d < DINNER; d += 256) {
    float s = bias[d];
#pragma unroll
    for (int k = 0; k < 4; ++k) {
      int t = l + k - 3;
      if (t >= 0)
        s += xz[(brow + t) * 2 * DINNER + d] * w[k * DINNER + d];
    }
    float val = s * (1.0f / (1.0f + __expf(-s)));   // silu
    u[(size_t)bl * DINNER + d]  = val;
    uh[(size_t)bl * DINNER + d] = (half_t)val;
  }
}

// =====================================================================
// delta = softplus(dt_proj_out + dt_proj_b)   (in place)
// =====================================================================
__global__ __launch_bounds__(256) void k_softplus_bias(
    float* __restrict__ delta, const float* __restrict__ bias, int n)
{
  int i = blockIdx.x * 256 + threadIdx.x;
  if (i >= n) return;
  float x = delta[i] + bias[i & (DINNER - 1)];
  delta[i] = (x > 20.0f) ? x : log1pf(__expf(x));
}

// =====================================================================
// Selective scan. One thread per (b,d) channel, N=16 state in registers.
// B_t / C_t (shared across d) staged in LDS per step. ys overwrites delta.
// =====================================================================
__global__ __launch_bounds__(256) void k_scan(
    const float* __restrict__ u,       // [B*L,1024]
    float* __restrict__ delta_ys,      // [B*L,1024] in: delta, out: ys
    const float* __restrict__ x_dbl,   // [B*L,64]  (cols 32..47 = B, 48..63 = C)
    const float* __restrict__ A_log)   // [1024,16]
{
  const int b = blockIdx.x >> 2;                  // 8 batches
  const int d = ((blockIdx.x & 3) << 8) + threadIdx.x;
  __shared__ float sB[DSTATE];
  __shared__ float sC[DSTATE];

  float a[DSTATE], h[DSTATE];
#pragma unroll
  for (int n = 0; n < DSTATE; ++n) {
    a[n] = -__expf(A_log[(size_t)d * DSTATE + n]);
    h[n] = 0.0f;
  }
  const size_t brow = (size_t)b * LL;
  for (int t = 0; t < LL; ++t) {
    __syncthreads();
    if (threadIdx.x < DSTATE)
      sB[threadIdx.x] = x_dbl[(brow + t) * 64 + 32 + threadIdx.x];
    else if (threadIdx.x < 2 * DSTATE)
      sC[threadIdx.x - DSTATE] = x_dbl[(brow + t) * 64 + 48 + (threadIdx.x - DSTATE)];
    __syncthreads();
    const size_t idx = (brow + t) * DINNER + d;
    const float ut = u[idx];
    const float dt = delta_ys[idx];
    const float du = dt * ut;
    float y = 0.0f;
#pragma unroll
    for (int n = 0; n < DSTATE; ++n) {
      h[n] = __expf(dt * a[n]) * h[n] + du * sB[n];
      y += h[n] * sC[n];
    }
    delta_ys[idx] = y;
  }
}

// =====================================================================
// y = (ys + u*D) * silu(z); emit f16 for out_proj GEMM
// =====================================================================
__global__ __launch_bounds__(256) void k_gate(
    const float* __restrict__ ys,     // [B*L,1024]
    const float* __restrict__ u,      // [B*L,1024]
    const float* __restrict__ xz,     // [B*L,2048] (z = cols 1024..2047)
    const float* __restrict__ Dp,     // [1024]
    half_t* __restrict__ yh, int n)
{
  int i = blockIdx.x * 256 + threadIdx.x;
  if (i >= n) return;
  int d  = i & (DINNER - 1);
  int bl = i >> 10;
  float z = xz[(size_t)bl * 2 * DINNER + DINNER + d];
  float y = (ys[i] + u[i] * Dp[d]) * (z / (1.0f + __expf(-z)));
  yh[i] = (half_t)y;
}

// =====================================================================
// Final projection: out[b,l,0] = dot(x_out[b,l,:512], proj_w) + proj_b
// =====================================================================
__global__ __launch_bounds__(256) void k_final(
    const float* __restrict__ xout,   // [B*L,512]
    const float* __restrict__ pw,     // [1,512]
    const float* __restrict__ pb,     // [1]
    float* __restrict__ out)          // [B*L]
{
  int i = blockIdx.x * 256 + threadIdx.x;
  if (i >= NTOK) return;
  float s = pb[0];
  const float* row = xout + (size_t)i * DMODEL;
#pragma unroll 8
  for (int k = 0; k < DMODEL; ++k) s += row[k] * pw[k];
  out[i] = s;
}

// =====================================================================
// Host launcher
// =====================================================================
extern "C" void kernel_launch(void* const* d_in, const int* in_sizes, int n_in,
                              void* d_out, int out_size, void* d_ws, size_t ws_size,
                              hipStream_t stream)
{
  const float* x_enc      = (const float*)d_in[0];
  const float* x_mark_enc = (const float*)d_in[1];
  const float* pos_emb    = (const float*)d_in[4];
  const float* conv_tok_w = (const float*)d_in[5];
  const float* temp_w     = (const float*)d_in[6];
  const float* in_proj_w  = (const float*)d_in[7];
  const float* conv1d_w   = (const float*)d_in[8];
  const float* conv1d_b   = (const float*)d_in[9];
  const float* x_proj_w   = (const float*)d_in[10];
  const float* dt_proj_w  = (const float*)d_in[11];
  const float* dt_proj_b  = (const float*)d_in[12];
  const float* A_log      = (const float*)d_in[13];
  const float* D_param    = (const float*)d_in[14];
  const float* out_proj_w = (const float*)d_in[15];
  const float* proj_w     = (const float*)d_in[16];
  const float* proj_b     = (const float*)d_in[17];
  float* out = (float*)d_out;

  // ---- carve workspace (256B aligned regions) ----
  char*  ws  = (char*)d_ws;
  size_t off = 0;
  auto carve = [&](size_t bytes) -> void* {
    off = (off + 255) & ~(size_t)255;
    void* p = ws + off;
    off += bytes;
    return p;
  };
  half_t* xh      = (half_t*)carve((size_t)NTOK * DMODEL * 2);       // embedded x (f16)
  float*  xz      = (float*) carve((size_t)NTOK * 2 * DINNER * 4);   // in_proj out
  float*  u       = (float*) carve((size_t)NTOK * DINNER * 4);       // post-conv SiLU
  half_t* uh      = (half_t*)carve((size_t)NTOK * DINNER * 2);       // u f16 (reused as y f16)
  float*  xdbl    = (float*) carve((size_t)NTOK * 64 * 4);           // x_proj out
  half_t* xdblh   = (half_t*)carve((size_t)NTOK * 64 * 2);
  float*  delta   = (float*) carve((size_t)NTOK * DINNER * 4);       // delta, then ys
  float*  xout    = (float*) carve((size_t)NTOK * DMODEL * 4);       // out_proj out
  half_t* w_in_h  = (half_t*)carve((size_t)2 * DINNER * DMODEL * 2);
  half_t* w_xp_h  = (half_t*)carve((size_t)64 * DINNER * 2);
  half_t* w_dt_h  = (half_t*)carve((size_t)DINNER * DTRANK * 2);
  half_t* w_out_h = (half_t*)carve((size_t)DMODEL * DINNER * 2);

  // ---- weight conversions ----
  auto cvt = [&](const float* s, half_t* d, int n) {
    k_cvt_f16<<<(n + 255) / 256, 256, 0, stream>>>(s, d, n);
  };
  cvt(in_proj_w,  w_in_h,  2 * DINNER * DMODEL);
  cvt(x_proj_w,   w_xp_h,  64 * DINNER);
  cvt(dt_proj_w,  w_dt_h,  DINNER * DTRANK);
  cvt(out_proj_w, w_out_h, DMODEL * DINNER);

  // ---- embedding -> xh ----
  k_embed<<<NTOK, 256, 0, stream>>>(x_enc, x_mark_enc, pos_emb, conv_tok_w, temp_w, xh);

  // ---- in_proj: [16384,512] x [2048,512]^T -> xz ----
  k_gemm_f16_wmma<<<dim3(NTOK / TILE_M, (2 * DINNER) / TILE_N), 256, 0, stream>>>(
      xh, DMODEL, w_in_h, xz, 2 * DINNER, NTOK, 2 * DINNER, DMODEL);

  // ---- depthwise conv + SiLU -> u, uh ----
  k_dwconv_silu<<<NTOK, 256, 0, stream>>>(xz, conv1d_w, conv1d_b, u, uh);

  // ---- x_proj: [16384,1024] x [64,1024]^T -> xdbl ----
  k_gemm_f16_wmma<<<dim3(NTOK / TILE_M, 64 / TILE_N), 256, 0, stream>>>(
      uh, DINNER, w_xp_h, xdbl, 64, NTOK, 64, DINNER);

  // ---- dt_proj: dt_in = xdbl[:, :32]; [16384,32] x [1024,32]^T -> delta ----
  cvt(xdbl, xdblh, NTOK * 64);
  k_gemm_f16_wmma<<<dim3(NTOK / TILE_M, DINNER / TILE_N), 256, 0, stream>>>(
      xdblh, 64, w_dt_h, delta, DINNER, NTOK, DINNER, DTRANK);

  // ---- softplus(delta + bias) ----
  k_softplus_bias<<<(NTOK * DINNER + 255) / 256, 256, 0, stream>>>(
      delta, dt_proj_b, NTOK * DINNER);

  // ---- selective scan (ys written over delta) ----
  k_scan<<<BB * (DINNER / 256), 256, 0, stream>>>(u, delta, xdbl, A_log);

  // ---- gate: y = (ys + u*D) * silu(z) -> f16 into uh ----
  k_gate<<<(NTOK * DINNER + 255) / 256, 256, 0, stream>>>(
      delta, u, xz, D_param, uh, NTOK * DINNER);

  // ---- out_proj: [16384,1024] x [512,1024]^T -> xout ----
  k_gemm_f16_wmma<<<dim3(NTOK / TILE_M, DMODEL / TILE_N), 256, 0, stream>>>(
      uh, DINNER, w_out_h, xout, DMODEL, NTOK, DMODEL, DINNER);

  // ---- final projection -> out ----
  k_final<<<(NTOK + 255) / 256, 256, 0, stream>>>(xout, proj_w, proj_b, out);
}
